// VanillaLSTM_79869211836499
// MI455X (gfx1250) — matrix operand order repaired and checked by reading, not compiled, not run on hardware
//
#include <hip/hip_runtime.h>
#include <hip/hip_bf16.h>

// ---- fixed problem sizes from the reference ----
#define BB   128          // batch
#define TT   256          // time steps
#define II   512          // input features
#define HH   1024         // hidden
#define CC   10           // classes
#define NG   4096         // 4*HH (gates f,i,c,o)
#define MM   (TT * BB)    // 32768 GEMM rows

typedef __attribute__((ext_vector_type(16))) _Float16 v16h;
typedef __attribute__((ext_vector_type(8)))  _Float16 v8h;
typedef __attribute__((ext_vector_type(8)))  float    v8f;

// LDS row stride (halves) for the weight tile: 1024 + 8 pad to kill bank conflicts
#define LDSW_STRIDE 1032
#define LDSW_BYTES  (64 * LDSW_STRIDE * 2)

// ---------------------------------------------------------------------------
// WMMA helpers (CDNA5 wave32, 16x16x32 f16 -> f32)
// ---------------------------------------------------------------------------
__device__ __forceinline__ v8f wmma16(v16h a, v16h b, v8f c) {
  // (neg_a, A, neg_b, B, c_mod, C, reuse_a, reuse_b)
  return __builtin_amdgcn_wmma_f32_16x16x32_f16(false, a, false, b, (short)0, c,
                                                false, false);
}

// A fragment: 16x32 f16, row-major source (row = M, ld = K stride).
// lanes 0-15: row, K[0:16] in halves[0:16]
// lanes16-31: row, K[16:32]
__device__ __forceinline__ v16h load_frag_a(const _Float16* __restrict__ base,
                                            int ld, int row0, int k, int lane) {
  int r  = row0 + (lane & 15);
  int ko = (lane >> 4) * 16;
  const _Float16* p = base + (size_t)r * ld + k + ko;
  union { v16h v; v8h h[2]; } u;
  u.h[0] = *(const v8h*)(p);
  u.h[1] = *(const v8h*)(p + 8);
  return u.v;
}

// B fragment: 32x16 f16 from an N-major (transposed) weight matrix WT[N][K].
// lane = column (N); lanes 0-15 hold K[0:16], lanes 16-31 hold K[16:32].
__device__ __forceinline__ v16h load_frag_b(const _Float16* __restrict__ baseT,
                                            int ld, int col0, int k, int lane) {
  int c  = col0 + (lane & 15);
  int ko = (lane >> 4) * 16;
  const _Float16* p = baseT + (size_t)c * ld + k + ko;
  union { v16h v; v8h h[2]; } u;
  u.h[0] = *(const v8h*)(p);
  u.h[1] = *(const v8h*)(p + 8);
  return u.v;
}

// Same B fragment but from the padded LDS weight tile (local row index r0).
__device__ __forceinline__ v16h load_frag_b_lds(const _Float16* sW, int r0,
                                                int k, int lane) {
  int r  = r0 + (lane & 15);
  int ko = (lane >> 4) * 16;
  const _Float16* p = sW + (size_t)r * LDSW_STRIDE + k + ko;
  union { v16h v; v8h h[2]; } u;
  u.h[0] = *(const v8h*)(p);
  u.h[1] = *(const v8h*)(p + 8);
  return u.v;
}

__device__ __forceinline__ float sigmoidf_(float x) {
  return 1.0f / (1.0f + __expf(-x));
}

// ---------------------------------------------------------------------------
// Prep kernels
// ---------------------------------------------------------------------------

// x (B,T,I) f32  ->  Xh (T*B, I) f16  (time-major rows)
__global__ __launch_bounds__(256) void prep_x_kernel(const float* __restrict__ x,
                                                     _Float16* __restrict__ Xh) {
  size_t idx = (size_t)blockIdx.x * blockDim.x + threadIdx.x;
  if (idx >= (size_t)MM * II) return;
  int    i = (int)(idx & (II - 1));
  size_t r = idx >> 9;               // II = 512
  int    t = (int)(r >> 7);          // BB = 128
  int    b = (int)(r & (BB - 1));
  Xh[idx] = (_Float16)x[((size_t)b * TT + t) * II + i];
}

// Gate weights (K,H) f32 x4  ->  WT (4H, K) f16, N-major, gate order f,i,c,o
__global__ __launch_bounds__(256) void prep_wT_kernel(const float* __restrict__ W0,
                                                      const float* __restrict__ W1,
                                                      const float* __restrict__ W2,
                                                      const float* __restrict__ W3,
                                                      _Float16* __restrict__ WT,
                                                      int K) {
  size_t idx = (size_t)blockIdx.x * blockDim.x + threadIdx.x;
  if (idx >= (size_t)NG * K) return;
  int k = (int)(idx % K);
  int n = (int)(idx / K);
  int g = n >> 10;
  int h = n & (HH - 1);
  const float* W = (g == 0) ? W0 : (g == 1) ? W1 : (g == 2) ? W2 : W3;
  WT[idx] = (_Float16)W[(size_t)k * HH + h];
}

// Fused bias, zero c / h0, reset barrier
__global__ __launch_bounds__(256) void prep_misc_kernel(const float* __restrict__ bfp,
                                                        const float* __restrict__ bip,
                                                        const float* __restrict__ bcp,
                                                        const float* __restrict__ bop,
                                                        float* __restrict__ bias,
                                                        float* __restrict__ c,
                                                        _Float16* __restrict__ h0buf,
                                                        unsigned* __restrict__ bar) {
  int idx = blockIdx.x * blockDim.x + threadIdx.x;
  if (idx < NG) {
    int g = idx >> 10, h = idx & (HH - 1);
    const float* s = (g == 0) ? bfp : (g == 1) ? bip : (g == 2) ? bcp : bop;
    bias[idx] = s[h];
  }
  if (idx < BB * HH) {
    c[idx]     = 0.0f;
    h0buf[idx] = (_Float16)0.0f;
  }
  if (idx < 2) bar[idx] = 0u;
}

// ---------------------------------------------------------------------------
// Input-projection GEMM: G(MM x NG) = Xh(MM x II) * WxT^T + bias, f16 out.
// 128x128 tile per WG; 8 waves, each 64x32 (4 M-tiles x 2 N-tiles).
// ---------------------------------------------------------------------------
__global__ __launch_bounds__(256) void proj_gemm_kernel(const _Float16* __restrict__ Xh,
                                                        const _Float16* __restrict__ WxT,
                                                        const float* __restrict__ bias,
                                                        _Float16* __restrict__ G) {
  const int lane = threadIdx.x & 31;
  const int wave = threadIdx.x >> 5;
  const int wm   = wave & 1;
  const int wn   = wave >> 1;
  const int mw   = blockIdx.y * 128 + wm * 64;
  const int nw   = blockIdx.x * 128 + wn * 32;

  const v8f z = {0.f, 0.f, 0.f, 0.f, 0.f, 0.f, 0.f, 0.f};
  v8f acc[4][2];
#pragma unroll
  for (int i = 0; i < 4; ++i)
#pragma unroll
    for (int j = 0; j < 2; ++j) acc[i][j] = z;

  for (int k = 0; k < II; k += 32) {
    v16h a[4], b[2];
#pragma unroll
    for (int i = 0; i < 4; ++i) a[i] = load_frag_a(Xh, II, mw + 16 * i, k, lane);
#pragma unroll
    for (int j = 0; j < 2; ++j) b[j] = load_frag_b(WxT, II, nw + 16 * j, k, lane);
#pragma unroll
    for (int i = 0; i < 4; ++i)
#pragma unroll
      for (int j = 0; j < 2; ++j) acc[i][j] = wmma16(a[i], b[j], acc[i][j]);
  }

#pragma unroll
  for (int j = 0; j < 2; ++j) {
    const int col = nw + 16 * j + (lane & 15);
    const float bz = bias[col];
#pragma unroll
    for (int i = 0; i < 4; ++i) {
      const int row0 = mw + 16 * i + (lane >> 4) * 8;
#pragma unroll
      for (int e = 0; e < 8; ++e)
        G[(size_t)(row0 + e) * NG + col] = (_Float16)(acc[i][j][e] + bz);
    }
  }
}

// ---------------------------------------------------------------------------
// Grid-wide sense-reversal barrier (persistent recurrent kernel)
// ---------------------------------------------------------------------------
__device__ __forceinline__ void grid_sync(unsigned* bar, unsigned nwg) {
  __threadfence();     // release this wave's h/c stores to device scope
  __syncthreads();
  if (threadIdx.x == 0) {
    unsigned g = __hip_atomic_load(&bar[1], __ATOMIC_RELAXED, __HIP_MEMORY_SCOPE_AGENT);
    unsigned a = __hip_atomic_fetch_add(&bar[0], 1u, __ATOMIC_ACQ_REL, __HIP_MEMORY_SCOPE_AGENT);
    if (a + 1 == nwg) {
      __hip_atomic_store(&bar[0], 0u, __ATOMIC_RELAXED, __HIP_MEMORY_SCOPE_AGENT);
      __hip_atomic_store(&bar[1], g + 1u, __ATOMIC_RELEASE, __HIP_MEMORY_SCOPE_AGENT);
    } else {
      while (__hip_atomic_load(&bar[1], __ATOMIC_ACQUIRE, __HIP_MEMORY_SCOPE_AGENT) == g)
        __builtin_amdgcn_s_sleep(1);
    }
  }
  __syncthreads();
  __threadfence();     // acquire: invalidate stale lines before reading new h
}

// ---------------------------------------------------------------------------
// Persistent recurrent kernel. 64 WGs; WG bx owns hidden cols [bx*16, bx*16+16)
// for ALL four gates, so gate fusion + c/h update are local. WhT slice (64 rows
// x 1024 K, 132 KB) lives in LDS for the whole sequence. One grid barrier/step.
// ---------------------------------------------------------------------------
__global__ __launch_bounds__(256) void lstm_recurrent_kernel(
    const _Float16* __restrict__ G,     // (TT, BB, NG) f16 pre-activations
    const _Float16* __restrict__ WhT,   // (NG, HH) f16 N-major
    _Float16* __restrict__ hbuf,        // 2 x (BB, HH) f16, ping-pong
    float* __restrict__ c,              // (BB, HH) f32 cell state
    float* __restrict__ hlast,          // (BB, HH) f32 final hidden
    unsigned* __restrict__ bar) {
  extern __shared__ _Float16 sW[];      // [64][LDSW_STRIDE]

  const int lane = threadIdx.x & 31;
  const int wave = threadIdx.x >> 5;
  const int h0   = blockIdx.x * 16;     // hidden column base
  const int m0   = wave * 16;           // batch-row base for this wave

  // Stage this WG's 64 weight rows (4 gates x 16 cols) into LDS, padded.
  for (int off = threadIdx.x * 8; off < 64 * HH; off += 256 * 8) {
    int r  = off >> 10;                 // local row 0..63 (HH = 1024)
    int kk = off & (HH - 1);
    int g  = r >> 4;
    int cc = r & 15;
    *(v8h*)&sW[(size_t)r * LDSW_STRIDE + kk] =
        *(const v8h*)&WhT[(size_t)(g * HH + h0 + cc) * HH + kk];
  }
  __syncthreads();

  const int rowl = m0 + (lane >> 4) * 8;   // C/D layout: row = rowl + e
  const int coll = lane & 15;

  for (int t = 0; t < TT; ++t) {
    const _Float16* hp = hbuf + (size_t)(t & 1) * (BB * HH);
    _Float16*       hn = hbuf + (size_t)((t + 1) & 1) * (BB * HH);
    const _Float16* Gt = G + (size_t)t * BB * NG;

    // Init accumulators from precomputed input projections (+bias already in).
    v8f acc[4];
#pragma unroll
    for (int g = 0; g < 4; ++g) {
      const int col = g * HH + h0 + coll;
#pragma unroll
      for (int e = 0; e < 8; ++e)
        acc[g][e] = (float)Gt[(size_t)(rowl + e) * NG + col];
    }

    // acc[g] += h_prev(16 rows) x WhT_tile(g)   over K = 1024
#pragma unroll 4
    for (int k = 0; k < HH; k += 32) {
      v16h a = load_frag_a(hp, HH, m0, k, lane);
#pragma unroll
      for (int g = 0; g < 4; ++g) {
        v16h b = load_frag_b_lds(sW, g * 16, k, lane);
        acc[g] = wmma16(a, b, acc[g]);
      }
    }

    // Gate nonlinearities + state update (f,i,c,o order).
#pragma unroll
    for (int e = 0; e < 8; ++e) {
      const int    brow = rowl + e;
      const int    hcol = h0 + coll;
      const size_t idx  = (size_t)brow * HH + hcol;
      float fg = sigmoidf_(acc[0][e]);
      float ig = sigmoidf_(acc[1][e]);
      float gg = tanhf(acc[2][e]);
      float og = sigmoidf_(acc[3][e]);
      float cn = fg * c[idx] + ig * gg;
      c[idx] = cn;
      float hv = og * tanhf(cn);
      hn[idx] = (_Float16)hv;
      if (t == TT - 1) hlast[idx] = hv;
    }

    grid_sync(bar, gridDim.x);
  }
}

// ---------------------------------------------------------------------------
// Classifier head: out(BB x CC) = hlast(BB x HH) @ fc_w(HH x CC) + fc_b
// ---------------------------------------------------------------------------
__global__ __launch_bounds__(128) void head_kernel(const float* __restrict__ hl,
                                                   const float* __restrict__ fcw,
                                                   const float* __restrict__ fcb,
                                                   float* __restrict__ out) {
  const int cls = blockIdx.x;   // 0..CC-1
  const int b   = threadIdx.x;  // 0..BB-1
  float s = fcb[cls];
  const float* hr = hl + (size_t)b * HH;
  for (int k = 0; k < HH; ++k) s += hr[k] * fcw[(size_t)k * CC + cls];
  out[(size_t)b * CC + cls] = s;
}

// ---------------------------------------------------------------------------
extern "C" void kernel_launch(void* const* d_in, const int* in_sizes, int n_in,
                              void* d_out, int out_size, void* d_ws, size_t ws_size,
                              hipStream_t stream) {
  const float* x   = (const float*)d_in[0];
  const float* Wxf = (const float*)d_in[1];
  const float* Whf = (const float*)d_in[2];
  const float* bf  = (const float*)d_in[3];
  const float* Wxi = (const float*)d_in[4];
  const float* Whi = (const float*)d_in[5];
  const float* bi  = (const float*)d_in[6];
  const float* Wxc = (const float*)d_in[7];
  const float* Whc = (const float*)d_in[8];
  const float* bc  = (const float*)d_in[9];
  const float* Wxo = (const float*)d_in[10];
  const float* Who = (const float*)d_in[11];
  const float* bo  = (const float*)d_in[12];
  const float* fcw = (const float*)d_in[13];
  const float* fcb = (const float*)d_in[14];
  float* out = (float*)d_out;

  // ---- workspace carve-out (256-byte aligned) ----
  char*  ws   = (char*)d_ws;
  size_t used = 0;
  auto alloc = [&](size_t bytes) -> char* {
    char* p = ws + used;
    used += (bytes + 255) & ~(size_t)255;
    return p;
  };
  _Float16* Xh    = (_Float16*)alloc((size_t)MM * II * 2);     //  33.6 MB
  _Float16* WxT   = (_Float16*)alloc((size_t)NG * II * 2);     //   4.2 MB
  _Float16* WhT   = (_Float16*)alloc((size_t)NG * HH * 2);     //   8.4 MB
  float*    bias  = (float*)   alloc((size_t)NG * 4);          //  16 KB
  _Float16* G     = (_Float16*)alloc((size_t)MM * NG * 2);     // 268.4 MB
  _Float16* hbuf  = (_Float16*)alloc((size_t)2 * BB * HH * 2); // 0.5 MB
  float*    cst   = (float*)   alloc((size_t)BB * HH * 4);     // 0.5 MB
  float*    hlast = (float*)   alloc((size_t)BB * HH * 4);     // 0.5 MB
  unsigned* bar   = (unsigned*)alloc(256);
  (void)ws_size; (void)in_sizes; (void)n_in; (void)out_size;

  // 1) prep: convert/transpose inputs & weights, fuse bias, zero state
  {
    size_t n = (size_t)MM * II;
    prep_x_kernel<<<(unsigned)((n + 255) / 256), 256, 0, stream>>>(x, Xh);
  }
  {
    size_t n = (size_t)NG * II;
    prep_wT_kernel<<<(unsigned)((n + 255) / 256), 256, 0, stream>>>(Wxf, Wxi, Wxc, Wxo, WxT, II);
  }
  {
    size_t n = (size_t)NG * HH;
    prep_wT_kernel<<<(unsigned)((n + 255) / 256), 256, 0, stream>>>(Whf, Whi, Whc, Who, WhT, HH);
  }
  prep_misc_kernel<<<(BB * HH + 255) / 256, 256, 0, stream>>>(bf, bi, bc, bo, bias,
                                                              cst, hbuf, bar);

  // 2) big WMMA GEMM: gate pre-activations for all timesteps
  {
    dim3 grid(NG / 128, MM / 128);  // 32 x 256
    proj_gemm_kernel<<<grid, 256, 0, stream>>>(Xh, WxT, bias, G);
  }

  // 3) persistent recurrence: 64 WGs, one grid barrier per timestep
  lstm_recurrent_kernel<<<HH / 16, 256, LDSW_BYTES, stream>>>(G, WhT, hbuf, cst,
                                                              hlast, bar);

  // 4) classifier head
  head_kernel<<<CC, BB, 0, stream>>>(hlast, fcw, fcb, out);
}
